// MixtureOfExperts_57217554317362
// MI455X (gfx1250) — compile-verified, matrix operand order checked by reading, not run on hardware
//
#include <hip/hip_runtime.h>
#include <hip/hip_bf16.h>
#include <math.h>

// Problem constants (from reference)
#define NTOK  2048      // B*S
#define DDIM  1024
#define HDIM  2048
#define NEXP  8
#define TOPK  2
#define ZCOEF 1e-4f
#define TOTSLOT (NTOK * TOPK + NTOK)   // 6144 hidden slots (routed + shared)

typedef __bf16          v16bf __attribute__((ext_vector_type(16)));
typedef unsigned short  v16u  __attribute__((ext_vector_type(16)));
typedef float           v8f   __attribute__((ext_vector_type(8)));
typedef unsigned int    u32x4 __attribute__((ext_vector_type(4)));
typedef int             i32x4 __attribute__((ext_vector_type(4)));
typedef int             i32x8 __attribute__((ext_vector_type(8)));

// 64 rows x (64 + 8 pad) ushorts: 144B row stride -> 16B aligned, no LDS bank
// conflicts on fragment loads, and expressible as TDM pad (4 DW per 32 DW).
#define LDS_STRIDE 72

__device__ __forceinline__ unsigned short f2bf(float f) {
  return __builtin_bit_cast(unsigned short, (__bf16)f);   // native v_cvt, RNE
}

__device__ __forceinline__ float silu_fast(float g) {
  // g * sigmoid(g) with v_rcp_f32 instead of IEEE divide (epilogue only,
  // result is immediately truncated to bf16).
  return g * __builtin_amdgcn_rcpf(1.0f + __expf(-g));
}

// 16x32 bf16 fragment from an LDS tile (ISA 7.12.2 striping); kk = 0 or 32.
__device__ __forceinline__ v16bf load_frag(const unsigned short* lds, int row0, int kk, int lane) {
  const unsigned short* p =
      lds + (row0 + (lane & 15)) * LDS_STRIDE + kk + ((lane >> 4) & 1) * 8;
  v16u t;
#pragma unroll
  for (int j = 0; j < 8; ++j) { t[j] = p[j]; t[j + 8] = p[j + 16]; }
  return __builtin_bit_cast(v16bf, t);
}

#define WMMA_BF16(A, B, C) \
  __builtin_amdgcn_wmma_f32_16x16x32_bf16(false, (A), false, (B), (short)0, (C), false, false)

// ---------------------------------------------------------------- zero
__global__ void moe_zero_kernel(float* out, int n, int* wsz) {
  int i = blockIdx.x * blockDim.x + threadIdx.x;
  if (i < n) out[i] = 0.0f;
  if (i < 64) wsz[i] = 0;   // counts[16], offs[16], stats[17] (+pad)
}

// ---------------------------------------------------------------- router
// One wave32 per token. stats: [0]=sum z^2, [1..8]=sum p_i, [9..16]=sum f_i
__global__ __launch_bounds__(256) void moe_router_kernel(
    const float* __restrict__ x, const float* __restrict__ rw, const float* __restrict__ rb,
    int* counts, int* list_tok, float* list_w, float* stats) {
  int wid = threadIdx.x >> 5, lane = threadIdx.x & 31;
  int n = blockIdx.x * 8 + wid;
  if (n >= NTOK) return;

  float acc[NEXP];
#pragma unroll
  for (int e = 0; e < NEXP; ++e) acc[e] = 0.0f;

  for (int d = lane; d < DDIM; d += 32) {
    float xv = x[(size_t)n * DDIM + d];
#pragma unroll
    for (int e = 0; e < NEXP; ++e) acc[e] += xv * rw[e * DDIM + d];
  }
#pragma unroll
  for (int e = 0; e < NEXP; ++e)
    for (int off = 16; off > 0; off >>= 1) acc[e] += __shfl_down(acc[e], off, 32);

  if (lane == 0) {
    float lg[NEXP];
#pragma unroll
    for (int e = 0; e < NEXP; ++e) lg[e] = acc[e];
    // biased top-2 (ties -> lowest index, matches jax top_k)
    float b0 = -1e30f; int i0 = 0;
#pragma unroll
    for (int e = 0; e < NEXP; ++e) { float be = lg[e] + rb[e]; if (be > b0) { b0 = be; i0 = e; } }
    float b1 = -1e30f; int i1 = (i0 == 0) ? 1 : 0;
#pragma unroll
    for (int e = 0; e < NEXP; ++e) if (e != i0) { float be = lg[e] + rb[e]; if (be > b1) { b1 = be; i1 = e; } }
    float l0 = lg[i0], l1 = lg[i1];
    float mx2 = fmaxf(l0, l1);
    float e0 = expf(l0 - mx2), e1 = expf(l1 - mx2);
    float inv = 1.0f / (e0 + e1);
    float w0 = e0 * inv, w1 = e1 * inv;
    float m = lg[0];
#pragma unroll
    for (int e = 1; e < NEXP; ++e) m = fmaxf(m, lg[e]);
    float s = 0.0f;
#pragma unroll
    for (int e = 0; e < NEXP; ++e) s += expf(lg[e] - m);
    float z = m + logf(s);
    atomicAdd(&stats[0], z * z);
#pragma unroll
    for (int e = 0; e < NEXP; ++e) atomicAdd(&stats[1 + e], expf(lg[e] - z));
    atomicAdd(&stats[9 + i0], 1.0f);
    atomicAdd(&stats[9 + i1], 1.0f);
    int p0 = atomicAdd(&counts[i0], 1);
    list_tok[i0 * NTOK + p0] = n; list_w[i0 * NTOK + p0] = w0;
    int p1 = atomicAdd(&counts[i1], 1);
    list_tok[i1 * NTOK + p1] = n; list_w[i1 * NTOK + p1] = w1;
  }
}

// ---------------------------------------------------------------- finalize
__global__ void moe_finalize_kernel(const int* counts, int* offs, const float* stats,
                                    float* out_tail) {
  if (threadIdx.x == 0) {
    int o = 0;
    for (int e = 0; e < NEXP; ++e) { offs[e] = o; o += counts[e]; }
    offs[NEXP] = o;  // shared-expert base (== NTOK*TOPK)
    float z_loss = stats[0] / (float)NTOK * ZCOEF;
    float lb = 0.0f;
    for (int e = 0; e < NEXP; ++e)
      lb += (stats[1 + e] / (float)NTOK) * (stats[9 + e] / (float)NTOK);
    lb *= (float)NEXP;
    out_tail[0] = 0.0f;    // aux_loss
    out_tail[1] = z_loss;  // router_z_loss
    out_tail[2] = lb;      // load_balance
  }
}

// ---------------------------------------------------------------- phase A
// hidden = silu(Xg @ G^T) * (Xg @ U^T), per-expert gathered tokens, bf16 out.
__global__ __launch_bounds__(256) void moe_gateup_kernel(
    const float* __restrict__ x,
    const float* __restrict__ gate_w, const float* __restrict__ up_w,
    const float* __restrict__ sh_gate_w, const float* __restrict__ sh_up_w,
    const int* __restrict__ counts, const int* __restrict__ offs,
    const int* __restrict__ list_tok, unsigned short* __restrict__ hidden) {
  int e = blockIdx.z;
  int cnt = (e < NEXP) ? counts[e] : NTOK;
  int m0 = blockIdx.y * 64;
  if (m0 >= cnt) return;
  int h0 = blockIdx.x * 64;
  const float* gptr = (e < NEXP) ? gate_w + (size_t)e * HDIM * DDIM : sh_gate_w;
  const float* uptr = (e < NEXP) ? up_w + (size_t)e * HDIM * DDIM : sh_up_w;
  int base = offs[(e < NEXP) ? e : NEXP];

  __shared__ unsigned short Xs[64 * LDS_STRIDE];
  __shared__ unsigned short Gs[64 * LDS_STRIDE];
  __shared__ unsigned short Us[64 * LDS_STRIDE];
  __shared__ int toks[64];

  int tid = threadIdx.x;
  if (tid < 64) {
    int mm = m0 + tid;
    toks[tid] = (e < NEXP) ? ((mm < cnt) ? list_tok[e * NTOK + mm] : 0) : mm;
  }
  __syncthreads();

  int lane = tid & 31, wid = tid >> 5;
  int mi = wid & 3;           // m sub-tile 0..3
  int nb = (wid >> 2) << 1;   // n sub-tile pair base: 0 or 2
  v8f cg0 = {}, cg1 = {}, cu0 = {}, cu1 = {};

  int lrow = tid >> 2;        // 0..63
  int lk = (tid & 3) * 8;     // 0,8,16,24

  for (int k0 = 0; k0 < DDIM; k0 += 64) {
    {
      const float* src = x + (size_t)toks[lrow] * DDIM + k0 + lk;
      unsigned short* dst = Xs + lrow * LDS_STRIDE + lk;
#pragma unroll
      for (int j = 0; j < 8; ++j) { dst[j] = f2bf(src[j]); dst[j + 32] = f2bf(src[j + 32]); }
    }
    {
      const float* src = gptr + (size_t)(h0 + lrow) * DDIM + k0 + lk;
      __builtin_prefetch(src + 64, 0, 1);   // next K chunk -> global_prefetch
      unsigned short* dst = Gs + lrow * LDS_STRIDE + lk;
#pragma unroll
      for (int j = 0; j < 8; ++j) { dst[j] = f2bf(src[j]); dst[j + 32] = f2bf(src[j + 32]); }
    }
    {
      const float* src = uptr + (size_t)(h0 + lrow) * DDIM + k0 + lk;
      __builtin_prefetch(src + 64, 0, 1);
      unsigned short* dst = Us + lrow * LDS_STRIDE + lk;
#pragma unroll
      for (int j = 0; j < 8; ++j) { dst[j] = f2bf(src[j]); dst[j + 32] = f2bf(src[j + 32]); }
    }
    __syncthreads();

    v16bf a0 = load_frag(Xs, mi * 16, 0, lane);
    v16bf a1 = load_frag(Xs, mi * 16, 32, lane);
    v16bf b;
    b = load_frag(Gs, nb * 16, 0, lane);        cg0 = WMMA_BF16(a0, b, cg0);
    b = load_frag(Gs, nb * 16, 32, lane);       cg0 = WMMA_BF16(a1, b, cg0);
    b = load_frag(Gs, (nb + 1) * 16, 0, lane);  cg1 = WMMA_BF16(a0, b, cg1);
    b = load_frag(Gs, (nb + 1) * 16, 32, lane); cg1 = WMMA_BF16(a1, b, cg1);
    b = load_frag(Us, nb * 16, 0, lane);        cu0 = WMMA_BF16(a0, b, cu0);
    b = load_frag(Us, nb * 16, 32, lane);       cu0 = WMMA_BF16(a1, b, cu0);
    b = load_frag(Us, (nb + 1) * 16, 0, lane);  cu1 = WMMA_BF16(a0, b, cu1);
    b = load_frag(Us, (nb + 1) * 16, 32, lane); cu1 = WMMA_BF16(a1, b, cu1);
    __syncthreads();
  }

  // epilogue: C layout — col = lane%16, row = i + 8*(lane/16)
  int rsub = (lane >> 4) * 8;
  int col = lane & 15;
#pragma unroll
  for (int i = 0; i < 8; ++i) {
    int gm = m0 + mi * 16 + i + rsub;
    if (gm < cnt) {
      float hv = silu_fast(cg0[i]) * cu0[i];
      hidden[(size_t)(base + gm) * HDIM + h0 + nb * 16 + col] = f2bf(hv);
      hv = silu_fast(cg1[i]) * cu1[i];
      hidden[(size_t)(base + gm) * HDIM + h0 + (nb + 1) * 16 + col] = f2bf(hv);
    }
  }
}

// ---------------------------------------------------------------- phase B
// out += w * (hidden @ Down^T); hidden tile DMA'd to LDS by the TDM with
// hardware row padding matching LDS_STRIDE (32 DW data + 4 DW pad per row).
__global__ __launch_bounds__(256) void moe_down_kernel(
    const float* __restrict__ down_w, const float* __restrict__ sh_down_w,
    const int* __restrict__ counts, const int* __restrict__ offs,
    const int* __restrict__ list_tok, const float* __restrict__ list_w,
    const unsigned short* __restrict__ hidden, float* __restrict__ out) {
  int e = blockIdx.z;
  int cnt = (e < NEXP) ? counts[e] : NTOK;
  int m0 = blockIdx.y * 64;
  if (m0 >= cnt) return;
  int d0 = blockIdx.x * 64;
  const float* dptr = (e < NEXP) ? down_w + (size_t)e * DDIM * HDIM : sh_down_w;
  int base = offs[(e < NEXP) ? e : NEXP];

  __shared__ unsigned short Hs[64 * LDS_STRIDE];
  __shared__ unsigned short Ds[64 * LDS_STRIDE];

  int tid = threadIdx.x;
  int lane = tid & 31, wid = tid >> 5;
  int mi = wid & 3;
  int nb = (wid >> 2) << 1;
  v8f c0 = {}, c1 = {};

  int lrow = tid >> 2;
  int lk = (tid & 3) * 8;

  // Invariant TDM descriptor parts (D#, ISA 8.3/8.4): 2D tensor of bf16,
  // 6144 x 2048, 64x64 tile, LDS pad 4 DW per 32 DW -> 144B row stride.
  unsigned lds_off = (unsigned)(unsigned long long)(size_t)(void*)Hs;
  unsigned long long gbase =
      (unsigned long long)(size_t)(const void*)hidden +
      ((size_t)(base + m0) * HDIM) * 2ull;

  for (int k0 = 0; k0 < HDIM; k0 += 64) {
    if (wid == 0) {
      unsigned long long ga = gbase + (unsigned long long)k0 * 2ull;
      u32x4 g0;
      g0[0] = 1u;                                   // count=1 (valid descriptor)
      g0[1] = lds_off;                              // lds_addr
      g0[2] = (unsigned)(ga & 0xFFFFFFFFu);         // global_addr[31:0]
      g0[3] = (unsigned)((ga >> 32) & 0x1FFFFFFu) | (2u << 30);  // addr[56:32] | type=2
      i32x8 g1;
      g1[0] = (int)((1u << 16)        // data_size = 2 bytes
                  | (1u << 20)        // pad_enable
                  | (4u << 22)        // pad_interval: 32 DWORDs
                  | (3u << 25));      // pad_amount : 4 DWORDs
      g1[1] = (int)(((unsigned)HDIM & 0xFFFFu) << 16);     // tensor_dim0 = 2048
      g1[2] = (int)(((unsigned)TOTSLOT & 0xFFFFu) << 16);  // tensor_dim1 = 6144
      g1[3] = (int)(64u << 16);                            // tile_dim0 = 64
      g1[4] = 64;                                          // tile_dim1 = 64
      g1[5] = HDIM;                                        // tensor_dim0_stride
      g1[6] = 0;
      g1[7] = 0;
      i32x4 z4 = {0, 0, 0, 0};
#if __clang_major__ >= 23
      i32x8 z8 = {0, 0, 0, 0, 0, 0, 0, 0};
      __builtin_amdgcn_tensor_load_to_lds(g0, g1, z4, z4, z8, 0);
#else
      __builtin_amdgcn_tensor_load_to_lds(g0, g1, z4, z4, 0);
#endif
    }
    {
      const float* src = dptr + (size_t)(d0 + lrow) * HDIM + k0 + lk;
      __builtin_prefetch(src + 64, 0, 1);
      unsigned short* dst = Ds + lrow * LDS_STRIDE + lk;
#pragma unroll
      for (int j = 0; j < 8; ++j) { dst[j] = f2bf(src[j]); dst[j + 32] = f2bf(src[j + 32]); }
    }
    if (wid == 0) __builtin_amdgcn_s_wait_tensorcnt(0);
    __syncthreads();

    v16bf a0 = load_frag(Hs, mi * 16, 0, lane);
    v16bf a1 = load_frag(Hs, mi * 16, 32, lane);
    v16bf b;
    b = load_frag(Ds, nb * 16, 0, lane);        c0 = WMMA_BF16(a0, b, c0);
    b = load_frag(Ds, nb * 16, 32, lane);       c0 = WMMA_BF16(a1, b, c0);
    b = load_frag(Ds, (nb + 1) * 16, 0, lane);  c1 = WMMA_BF16(a0, b, c1);
    b = load_frag(Ds, (nb + 1) * 16, 32, lane); c1 = WMMA_BF16(a1, b, c1);
    __syncthreads();
  }

  int rsub = (lane >> 4) * 8;
  int col = lane & 15;
#pragma unroll
  for (int i = 0; i < 8; ++i) {
    int gm = m0 + mi * 16 + i + rsub;
    if (gm < cnt) {
      int tok; float w;
      if (e < NEXP) { tok = list_tok[e * NTOK + gm]; w = list_w[e * NTOK + gm]; }
      else          { tok = gm; w = 1.0f; }
      atomicAdd(&out[(size_t)tok * DDIM + d0 + nb * 16 + col], w * c0[i]);
      atomicAdd(&out[(size_t)tok * DDIM + d0 + (nb + 1) * 16 + col], w * c1[i]);
    }
  }
}

// ---------------------------------------------------------------- launch
extern "C" void kernel_launch(void* const* d_in, const int* in_sizes, int n_in,
                              void* d_out, int out_size, void* d_ws, size_t ws_size,
                              hipStream_t stream) {
  (void)in_sizes; (void)n_in; (void)out_size; (void)ws_size;
  const float* x         = (const float*)d_in[0];
  const float* router_w  = (const float*)d_in[1];
  const float* router_b  = (const float*)d_in[2];
  const float* gate_w    = (const float*)d_in[3];
  const float* up_w      = (const float*)d_in[4];
  const float* down_w    = (const float*)d_in[5];
  const float* sh_gate_w = (const float*)d_in[6];
  const float* sh_up_w   = (const float*)d_in[7];
  const float* sh_down_w = (const float*)d_in[8];
  float* out = (float*)d_out;

  // workspace layout
  char* ws = (char*)d_ws;
  int*   counts   = (int*)ws;                       // 16 ints
  int*   offs     = (int*)(ws + 64);                // 16 ints
  float* stats    = (float*)(ws + 128);             // 17 floats (z2, p[8], f[8])
  int*   list_tok = (int*)(ws + 256);               // 8*2048 ints
  float* list_w   = (float*)(ws + 256 + NEXP * NTOK * 4);
  unsigned short* hidden =
      (unsigned short*)(ws + 256 + 2 * NEXP * NTOK * 4);  // 6144 * 2048 bf16 (~25 MB)

  const int ND = NTOK * DDIM;
  moe_zero_kernel<<<(ND + 255) / 256, 256, 0, stream>>>(out, ND, (int*)ws);
  moe_router_kernel<<<NTOK / 8, 256, 0, stream>>>(x, router_w, router_b,
                                                  counts, list_tok, list_w, stats);
  moe_finalize_kernel<<<1, 32, 0, stream>>>(counts, offs, stats, out + ND);
  moe_gateup_kernel<<<dim3(HDIM / 64, NTOK / 64, NEXP + 1), 256, 0, stream>>>(
      x, gate_w, up_w, sh_gate_w, sh_up_w, counts, offs, list_tok, hidden);
  moe_down_kernel<<<dim3(DDIM / 64, NTOK / 64, NEXP + 1), 256, 0, stream>>>(
      down_w, sh_down_w, counts, offs, list_tok, list_w, hidden, out);
}
